// ECGMamba_64364379898454
// MI455X (gfx1250) — compile-verified, matrix operand order checked by reading, not run on hardware
//
#include <hip/hip_runtime.h>
#include <hip/hip_bf16.h>
#include <math.h>

// ---------------------------------------------------------------------------
// ECG-Mamba forward on MI455X (gfx1250, wave32, WMMA + TDM).
//  - Projections via v_wmma_f32_16x16x32_f16; weight strips staged into LDS
//    once per 8-wave block with TENSOR_LOAD_TO_LDS (TDM) + s_wait_tensorcnt.
//  - Multi-accumulator waves, wave-uniform epilogues.
//  - Scan stays f32, gating fused, global_prefetch_b8 ahead of the recurrence.
// ---------------------------------------------------------------------------

#define Bsz     16
#define Lseq    2048
#define DMODEL  128
#define DINNER  256
#define DSTATE  16
#define DTRANK  8
#define DCONV   4
#define NLAYERS 4
#define NCLS    5
#define MTOK    (Bsz * Lseq)          // 32768 tokens
#define MTILES  (MTOK / 16)           // 2048 M-tiles
#define WPB     8                     // waves per GEMM block

typedef __attribute__((ext_vector_type(16))) _Float16 v16h;
typedef __attribute__((ext_vector_type(8)))  _Float16 v8h;
typedef __attribute__((ext_vector_type(8)))  float    v8f;
typedef __attribute__((ext_vector_type(4)))  unsigned int v4u;
typedef __attribute__((ext_vector_type(8)))  int      v8i;
typedef __attribute__((ext_vector_type(4)))  int      v4i;

#define WMMA_F16(a, b, c) \
  __builtin_amdgcn_wmma_f32_16x16x32_f16(false, (a), false, (b), (short)0, (c), false, false)

// ---------------- TDM: stage a rows x width f16 tile (row-major) into LDS ----
// D# per CDNA5 ISA: group0 {count, lds_addr, global_addr,
// type=2}; group1 {data_size=2B, tensor_dim0=width, tensor_dim1=rows,
// tile_dim0=width, tile_dim1=rows, dim0_stride=width}; groups 2/3 zero (2-D).
// Toolchain here is the 6-arg builtin: (v4u, v8i, v4i, v4i, v8i, i32 cpol).
__device__ __forceinline__ void tdm_load_2d_f16(const _Float16* gsrc,
                                                _Float16* lds_dst,
                                                int width, int rows) {
  unsigned long long ga = (unsigned long long)(const void*)gsrc;
  unsigned int la = (unsigned int)(unsigned long long)(void*)lds_dst; // low 32b = LDS offset
  v4u g0;
  g0[0] = 1u;                                              // count=1 (valid user D#)
  g0[1] = la;                                              // lds_addr (bytes)
  g0[2] = (unsigned int)(ga & 0xFFFFFFFFu);                // global_addr[31:0]
  g0[3] = (unsigned int)((ga >> 32) & 0x1FFFFFFu)          // global_addr[56:32]
        | (2u << 30);                                      // type=2 ("image")
  v8i g1;
  g1[0] = (1 << 16);                                       // data_size=1 -> 2 bytes
  g1[1] = (width & 0xFFFF) << 16;                          // tensor_dim0[15:0]
  g1[2] = ((width >> 16) & 0xFFFF) | ((rows & 0xFFFF) << 16); // dim0 hi | dim1 lo
  g1[3] = ((rows >> 16) & 0xFFFF) | ((width & 0xFFFF) << 16); // dim1 hi | tile_dim0
  g1[4] = rows & 0xFFFF;                                   // tile_dim1 (tile_dim2=0)
  g1[5] = width;                                           // tensor_dim0_stride[31:0]
  g1[6] = 0;                                               // stride hi | dim1_stride lo
  g1[7] = 0;
  v4i z4 = {};
  v8i z8 = {};
  __builtin_amdgcn_tensor_load_to_lds(g0, g1, z4, z4, z8, 0);
}

// ---------------- helpers ----------------

__global__ void k_f32_to_f16(const float* __restrict__ src,
                             _Float16* __restrict__ dst, int n) {
  int t = blockIdx.x * blockDim.x + threadIdx.x;
  if (t < n) dst[t] = (_Float16)src[t];
}

// h = x @ enc_w^T + enc_b ; one thread per (token, d)
__global__ void k_encoder(const float* __restrict__ x,
                          const float* __restrict__ ew,
                          const float* __restrict__ eb,
                          float* __restrict__ h) {
  int t = blockIdx.x * blockDim.x + threadIdx.x;   // MTOK*128
  int tok = t >> 7;
  int d = t & 127;
  const float* xp = x + (size_t)tok * 12;
  const float* wp = ew + d * 12;
  float acc = eb[d];
#pragma unroll
  for (int c = 0; c < 12; ++c) acc = fmaf(xp[c], wp[c], acc);
  h[t] = acc;
}

// rmsnorm per token (blockDim=128), emit f16 for the WMMA consumer
__global__ void k_rmsnorm_f16(const float* __restrict__ h,
                              const float* __restrict__ w,
                              _Float16* __restrict__ out) {
  __shared__ float red[DMODEL];
  int tok = blockIdx.x;
  int d = threadIdx.x;
  float v = h[(size_t)tok * DMODEL + d];
  red[d] = v * v;
  __syncthreads();
  for (int s = DMODEL / 2; s > 0; s >>= 1) {
    if (d < s) red[d] += red[d + s];
    __syncthreads();
  }
  float inv = rsqrtf(red[0] * (1.0f / DMODEL) + 1e-5f);
  out[(size_t)tok * DMODEL + d] = (_Float16)(v * inv * w[d]);
}

// ---------------- WMMA fragment load ----------------
// 16x32 f16 fragment: lane<16 -> row=lane,    k in {0..7,16..23}
//                     lane>=16 -> row=lane-16, k in {8..15,24..31}
__device__ __forceinline__ v16h ld16x32(const _Float16* __restrict__ row, int ksel) {
  v8h lo = *(const v8h*)(row + ksel);
  v8h hi = *(const v8h*)(row + ksel + 16);
  v16h o;
#pragma unroll
  for (int i = 0; i < 8; ++i) { o[i] = lo[i]; o[i + 8] = hi[i]; }
  return o;
}

// in_proj: Y[M x 512] = hn16[M x 128] * W[512 x 128]^T
// Block = 8 waves; the 64x128 weight strip is TDM-staged into LDS once and
// shared by all waves (each wave owns one M-tile, 4 accumulators).
// grid = (MTILES/8, 8). tg<4 -> xm (f32); tg>=4 -> z16 (f16) [wave-uniform].
__global__ void k_inproj(const _Float16* __restrict__ hn16,
                         const _Float16* __restrict__ w16,  // 512 x 128
                         float* __restrict__ xm,            // M x 256
                         _Float16* __restrict__ z16) {      // M x 256
  __shared__ _Float16 shB[64 * DMODEL];                     // 16 KB
  int tg = blockIdx.y;
  if (threadIdx.x < 32) {
    tdm_load_2d_f16(w16 + (size_t)(tg * 64) * DMODEL, shB, DMODEL, 64);
    __builtin_amdgcn_s_wait_tensorcnt(0);
  }
  __syncthreads();

  int wv = threadIdx.x >> 5;
  int lane = threadIdx.x & 31;
  int tm = blockIdx.x * WPB + wv;
  int r = lane & 15;
  int ksel = (lane < 16) ? 0 : 8;
  const _Float16* arow = hn16 + (size_t)(tm * 16 + r) * DMODEL;
  const _Float16* brow = shB + (size_t)r * DMODEL;
  v8f acc0 = {}, acc1 = {}, acc2 = {}, acc3 = {};
#pragma unroll
  for (int k = 0; k < DMODEL; k += 32) {
    v16h a  = ld16x32(arow + k, ksel);
    v16h b0 = ld16x32(brow + 0 * 16 * DMODEL + k, ksel);
    v16h b1 = ld16x32(brow + 1 * 16 * DMODEL + k, ksel);
    v16h b2 = ld16x32(brow + 2 * 16 * DMODEL + k, ksel);
    v16h b3 = ld16x32(brow + 3 * 16 * DMODEL + k, ksel);
    acc0 = WMMA_F16(a, b0, acc0);
    acc1 = WMMA_F16(a, b1, acc1);
    acc2 = WMMA_F16(a, b2, acc2);
    acc3 = WMMA_F16(a, b3, acc3);
  }
  int mbase = tm * 16 + ((lane < 16) ? 0 : 8);
  if (tg < 4) {                       // uniform across the wave
    int nb = tg * 64 + r;
#pragma unroll
    for (int rr = 0; rr < 8; ++rr) {
      size_t row = (size_t)(mbase + rr) * DINNER;
      xm[row + nb +  0] = acc0[rr];
      xm[row + nb + 16] = acc1[rr];
      xm[row + nb + 32] = acc2[rr];
      xm[row + nb + 48] = acc3[rr];
    }
  } else {
    int nb = (tg - 4) * 64 + r;
#pragma unroll
    for (int rr = 0; rr < 8; ++rr) {
      size_t row = (size_t)(mbase + rr) * DINNER;
      z16[row + nb +  0] = (_Float16)acc0[rr];
      z16[row + nb + 16] = (_Float16)acc1[rr];
      z16[row + nb + 32] = (_Float16)acc2[rr];
      z16[row + nb + 48] = (_Float16)acc3[rr];
    }
  }
}

// x_proj: xdbl[M x 40] = xc16[M x 256] * W[40 x 256]^T, N padded to 48.
// 40x256 weight strip TDM-staged into LDS; 8 waves x 1 M-tile, 3 accumulators.
// grid = (MTILES/8).
__global__ void k_xproj(const _Float16* __restrict__ xc16,
                        const _Float16* __restrict__ w16,  // 40 x 256
                        float* __restrict__ xdbl) {        // M x 40
  __shared__ _Float16 shB[40 * DINNER];                    // 20 KB
  if (threadIdx.x < 32) {
    tdm_load_2d_f16(w16, shB, DINNER, 40);
    __builtin_amdgcn_s_wait_tensorcnt(0);
  }
  __syncthreads();

  int wv = threadIdx.x >> 5;
  int lane = threadIdx.x & 31;
  int tm = blockIdx.x * WPB + wv;
  int r = lane & 15;
  int ksel = (lane < 16) ? 0 : 8;
  const _Float16* arow = xc16 + (size_t)(tm * 16 + r) * DINNER;
  const _Float16* b0r = shB + (size_t)(0 + r) * DINNER;
  const _Float16* b1r = shB + (size_t)(16 + r) * DINNER;
  const _Float16* b2r = shB + (size_t)(32 + r) * DINNER;
  bool b2valid = (32 + r) < (DTRANK + 2 * DSTATE);   // rows 32..39 only
  v8f acc0 = {}, acc1 = {}, acc2 = {};
#pragma unroll
  for (int k = 0; k < DINNER; k += 32) {
    v16h a  = ld16x32(arow + k, ksel);
    v16h b0 = ld16x32(b0r + k, ksel);
    v16h b1 = ld16x32(b1r + k, ksel);
    v16h b2 = {};
    if (b2valid) b2 = ld16x32(b2r + k, ksel);
    acc0 = WMMA_F16(a, b0, acc0);
    acc1 = WMMA_F16(a, b1, acc1);
    acc2 = WMMA_F16(a, b2, acc2);
  }
  int mbase = tm * 16 + ((lane < 16) ? 0 : 8);
#pragma unroll
  for (int rr = 0; rr < 8; ++rr) {
    size_t row = (size_t)(mbase + rr) * 40;
    xdbl[row + r]      = acc0[rr];
    xdbl[row + 16 + r] = acc1[rr];
    if (b2valid) xdbl[row + 32 + r] = acc2[rr];
  }
}

// out_proj: h[M x 128] += yg16[M x 256] * W[128 x 256]^T (residual fused)
// 64x256 weight strip TDM-staged into LDS; 8 waves x 1 M-tile, 4 accumulators.
// grid = (MTILES/8, 2).
__global__ void k_outproj(const _Float16* __restrict__ yg16,
                          const _Float16* __restrict__ w16,  // 128 x 256
                          float* __restrict__ h) {
  __shared__ _Float16 shB[64 * DINNER];                      // 32 KB
  int tg = blockIdx.y;
  if (threadIdx.x < 32) {
    tdm_load_2d_f16(w16 + (size_t)(tg * 64) * DINNER, shB, DINNER, 64);
    __builtin_amdgcn_s_wait_tensorcnt(0);
  }
  __syncthreads();

  int wv = threadIdx.x >> 5;
  int lane = threadIdx.x & 31;
  int tm = blockIdx.x * WPB + wv;
  int r = lane & 15;
  int ksel = (lane < 16) ? 0 : 8;
  const _Float16* arow = yg16 + (size_t)(tm * 16 + r) * DINNER;
  const _Float16* brow = shB + (size_t)r * DINNER;
  v8f acc0 = {}, acc1 = {}, acc2 = {}, acc3 = {};
#pragma unroll
  for (int k = 0; k < DINNER; k += 32) {
    v16h a  = ld16x32(arow + k, ksel);
    v16h b0 = ld16x32(brow + 0 * 16 * DINNER + k, ksel);
    v16h b1 = ld16x32(brow + 1 * 16 * DINNER + k, ksel);
    v16h b2 = ld16x32(brow + 2 * 16 * DINNER + k, ksel);
    v16h b3 = ld16x32(brow + 3 * 16 * DINNER + k, ksel);
    acc0 = WMMA_F16(a, b0, acc0);
    acc1 = WMMA_F16(a, b1, acc1);
    acc2 = WMMA_F16(a, b2, acc2);
    acc3 = WMMA_F16(a, b3, acc3);
  }
  int mbase = tm * 16 + ((lane < 16) ? 0 : 8);
  int nb = tg * 64 + r;
#pragma unroll
  for (int rr = 0; rr < 8; ++rr) {
    size_t row = (size_t)(mbase + rr) * DMODEL;
    h[row + nb +  0] += acc0[rr];
    h[row + nb + 16] += acc1[rr];
    h[row + nb + 32] += acc2[rr];
    h[row + nb + 48] += acc3[rr];
  }
}

// ---------------- conv + SiLU (writes f32 for scan-u, f16 for x_proj) ----------------
__global__ void k_conv_silu(const float* __restrict__ xm,
                            const float* __restrict__ cw,  // 256 x 4
                            const float* __restrict__ cb,  // 256
                            float* __restrict__ xc,
                            _Float16* __restrict__ xc16) {
  int t = blockIdx.x * blockDim.x + threadIdx.x;   // MTOK*256
  int d = t & 255;
  int tok = t >> 8;
  int b = tok / Lseq;
  int l = tok - b * Lseq;
  float acc = cb[d];
#pragma unroll
  for (int j = 0; j < DCONV; ++j) {
    int ls = l + j - (DCONV - 1);
    if (ls >= 0)
      acc = fmaf(cw[d * DCONV + j], xm[((size_t)b * Lseq + ls) * DINNER + d], acc);
  }
  float s = acc / (1.0f + __expf(-acc));
  xc[t] = s;
  xc16[t] = (_Float16)s;
}

// ---------------- delta = softplus(dt @ dt_w^T + dt_b) ----------------
__global__ void k_delta(const float* __restrict__ xdbl,    // M x 40 (dt = [0..8))
                        const float* __restrict__ dtw,     // 256 x 8
                        const float* __restrict__ dtb,     // 256
                        float* __restrict__ delta) {       // M x 256
  int t = blockIdx.x * blockDim.x + threadIdx.x;   // MTOK*256
  int d = t & 255;
  int tok = t >> 8;
  const float* xp = xdbl + (size_t)tok * 40;
  float acc = dtb[d];
#pragma unroll
  for (int r = 0; r < DTRANK; ++r) acc = fmaf(xp[r], dtw[d * DTRANK + r], acc);
  delta[t] = (acc > 20.0f) ? acc : log1pf(__expf(acc));
}

// ---------------- selective scan, gating fused ----------------
// 16 lanes per (b,d) recurrence, one state per lane; half-wave shfl_xor reduce.
// Prefetch ~8 tokens ahead (global_prefetch_b8) to hide latency behind v_exp.
__global__ void k_scan(const float* __restrict__ delta,    // M x 256
                       const float* __restrict__ u,        // M x 256 (conv+silu, f32)
                       const float* __restrict__ xdbl,     // M x 40 (B=[8..24), C=[24..40))
                       const _Float16* __restrict__ z16,   // M x 256
                       const float* __restrict__ A_log,    // 256 x 16 (layer slice)
                       const float* __restrict__ Dp,       // 256 (layer slice)
                       _Float16* __restrict__ yg16) {      // M x 256 out (gated, f16)
  int t = blockIdx.x * blockDim.x + threadIdx.x;  // 4096 pairs * 16
  int pair = t >> 4;
  int n = t & 15;
  int b = pair >> 8;
  int d = pair & 255;
  float a = -__expf(A_log[d * DSTATE + n]);
  float Dv = Dp[d];
  float hst = 0.0f;
  size_t base = (size_t)b * Lseq;
  for (int l = 0; l < Lseq; ++l) {
    size_t tok = base + l;
    if (l + 8 < Lseq) {
      size_t ftok = tok + 8;
      __builtin_prefetch(&delta[ftok * DINNER + d], 0, 0);
      __builtin_prefetch(&u[ftok * DINNER + d], 0, 0);
      __builtin_prefetch(&xdbl[ftok * 40 + DTRANK + n], 0, 0);
    }
    float dv = delta[tok * DINNER + d];
    float uv = u[tok * DINNER + d];
    float Bn = xdbl[tok * 40 + DTRANK + n];
    float Cn = xdbl[tok * 40 + DTRANK + DSTATE + n];
    float dA = __expf(dv * a);
    hst = fmaf(dA, hst, dv * Bn * uv);
    float p = hst * Cn;
    p += __shfl_xor(p, 1);
    p += __shfl_xor(p, 2);
    p += __shfl_xor(p, 4);
    p += __shfl_xor(p, 8);
    if (n == 0) {
      float y = fmaf(uv, Dv, p);
      float zv = (float)z16[tok * DINNER + d];
      float sz = zv / (1.0f + __expf(-zv));
      yg16[tok * DINNER + d] = (_Float16)(y * sz);
    }
  }
}

// ---------------- final norm/pool/classify ----------------
__global__ void k_token_inv(const float* __restrict__ h, float* __restrict__ inv) {
  __shared__ float red[DMODEL];
  int tok = blockIdx.x;
  int d = threadIdx.x;
  float v = h[(size_t)tok * DMODEL + d];
  red[d] = v * v;
  __syncthreads();
  for (int s = DMODEL / 2; s > 0; s >>= 1) {
    if (d < s) red[d] += red[d + s];
    __syncthreads();
  }
  if (d == 0) inv[tok] = rsqrtf(red[0] * (1.0f / DMODEL) + 1e-5f);
}

__global__ void k_pool(const float* __restrict__ h, const float* __restrict__ inv,
                       const float* __restrict__ normw, float* __restrict__ pooled) {
  int b = blockIdx.x;
  int d = threadIdx.x;
  float acc = 0.0f;
  for (int l = 0; l < Lseq; ++l) {
    size_t tok = (size_t)b * Lseq + l;
    acc = fmaf(h[tok * DMODEL + d], inv[tok], acc);
  }
  pooled[b * DMODEL + d] = acc * normw[d] * (1.0f / Lseq);
}

__global__ void k_cls(const float* __restrict__ pooled, const float* __restrict__ cw,
                      const float* __restrict__ cb, float* __restrict__ out) {
  int i = threadIdx.x;
  if (i >= Bsz * NCLS) return;
  int b = i / NCLS, c = i - b * NCLS;
  float acc = cb[c];
#pragma unroll 8
  for (int d = 0; d < DMODEL; ++d) acc = fmaf(pooled[b * DMODEL + d], cw[c * DMODEL + d], acc);
  out[b * NCLS + c] = acc;
}

// ---------------- host launcher ----------------
extern "C" void kernel_launch(void* const* d_in, const int* in_sizes, int n_in,
                              void* d_out, int out_size, void* d_ws, size_t ws_size,
                              hipStream_t stream) {
  (void)in_sizes; (void)n_in; (void)out_size; (void)ws_size;
  const float* x        = (const float*)d_in[0];
  const float* enc_w    = (const float*)d_in[1];
  const float* enc_b    = (const float*)d_in[2];
  const float* norm_w   = (const float*)d_in[3];
  const float* in_w     = (const float*)d_in[4];
  const float* conv_w   = (const float*)d_in[5];
  const float* conv_b   = (const float*)d_in[6];
  const float* xp_w     = (const float*)d_in[7];
  const float* dt_w     = (const float*)d_in[8];
  const float* dt_b     = (const float*)d_in[9];
  const float* A_log    = (const float*)d_in[10];
  const float* Dp       = (const float*)d_in[11];
  const float* out_w    = (const float*)d_in[12];
  const float* norm_f   = (const float*)d_in[13];
  const float* cls_w    = (const float*)d_in[14];
  const float* cls_b    = (const float*)d_in[15];

  char* ws = (char*)d_ws;
  size_t off = 0;
  auto carve = [&](size_t bytes) -> char* {
    char* p = ws + off;
    off += (bytes + 255) & ~(size_t)255;
    return p;
  };
  float*    h      = (float*)   carve((size_t)MTOK * DMODEL * 4);
  _Float16* hn16   = (_Float16*)carve((size_t)MTOK * DMODEL * 2);
  float*    xm     = (float*)   carve((size_t)MTOK * DINNER * 4);   // later reused as delta
  _Float16* z16    = (_Float16*)carve((size_t)MTOK * DINNER * 2);
  float*    xc     = (float*)   carve((size_t)MTOK * DINNER * 4);
  _Float16* xc16   = (_Float16*)carve((size_t)MTOK * DINNER * 2);
  float*    xdbl   = (float*)   carve((size_t)MTOK * 40 * 4);
  _Float16* yg16   = (_Float16*)carve((size_t)MTOK * DINNER * 2);
  _Float16* w16in  = (_Float16*)carve((size_t)NLAYERS * 2 * DINNER * DMODEL * 2);
  _Float16* w16xp  = (_Float16*)carve((size_t)NLAYERS * 40 * DINNER * 2);
  _Float16* w16out = (_Float16*)carve((size_t)NLAYERS * DMODEL * DINNER * 2);
  float*    invb   = (float*)   carve((size_t)MTOK * 4);
  float*    pooled = (float*)   carve((size_t)Bsz * DMODEL * 4);
  float*    deltaB = xm;  // reuse: conv fully consumes xm before k_delta writes here

  // weight conversion (deterministic, every call)
  int nin  = NLAYERS * 2 * DINNER * DMODEL;
  int nxp  = NLAYERS * 40 * DINNER;
  int nout = NLAYERS * DMODEL * DINNER;
  k_f32_to_f16<<<(nin  + 255) / 256, 256, 0, stream>>>(in_w,  w16in,  nin);
  k_f32_to_f16<<<(nxp  + 255) / 256, 256, 0, stream>>>(xp_w,  w16xp,  nxp);
  k_f32_to_f16<<<(nout + 255) / 256, 256, 0, stream>>>(out_w, w16out, nout);

  // encoder
  k_encoder<<<(MTOK * DMODEL) / 256, 256, 0, stream>>>(x, enc_w, enc_b, h);

  for (int i = 0; i < NLAYERS; ++i) {
    k_rmsnorm_f16<<<MTOK, DMODEL, 0, stream>>>(h, norm_w + i * DMODEL, hn16);
    k_inproj<<<dim3(MTILES / WPB, 8), 32 * WPB, 0, stream>>>(
        hn16, w16in + (size_t)i * 2 * DINNER * DMODEL, xm, z16);
    k_conv_silu<<<(MTOK * DINNER) / 256, 256, 0, stream>>>(
        xm, conv_w + i * DINNER * DCONV, conv_b + i * DINNER, xc, xc16);
    k_xproj<<<MTILES / WPB, 32 * WPB, 0, stream>>>(
        xc16, w16xp + (size_t)i * 40 * DINNER, xdbl);
    k_delta<<<(MTOK * DINNER) / 256, 256, 0, stream>>>(
        xdbl, dt_w + i * DINNER * DTRANK, dt_b + i * DINNER, deltaB);
    k_scan<<<(Bsz * DINNER * DSTATE) / 256, 256, 0, stream>>>(
        deltaB, xc, xdbl, z16, A_log + (size_t)i * DINNER * DSTATE,
        Dp + i * DINNER, yg16);
    k_outproj<<<dim3(MTILES / WPB, 2), 32 * WPB, 0, stream>>>(
        yg16, w16out + (size_t)i * DMODEL * DINNER, h);
  }

  k_token_inv<<<MTOK, DMODEL, 0, stream>>>(h, invb);
  k_pool<<<Bsz, DMODEL, 0, stream>>>(h, invb, norm_f, pooled);
  k_cls<<<1, 128, 0, stream>>>(pooled, cls_w, cls_b, (float*)d_out);
}
